// Encoder_13159779795615
// MI455X (gfx1250) — compile-verified, matrix operand order checked by reading, not run on hardware
//
#include <hip/hip_runtime.h>
#include <hip/hip_bf16.h>
#include <math.h>

// Problem constants (match reference)
#define B_   16
#define N_   400
#define P_   200
#define ED_  200   // E_DIM
#define AD_  128   // A_DIM
#define H_   8
#define EPAD 224   // E_DIM padded to multiple of 32 (WMMA K)
#define NPAD 416   // N padded to multiple of 32 (WMMA K for pooling)
#define PPAD 208   // P padded to multiple of 16 (A-operand rows for pooling)

typedef __attribute__((ext_vector_type(16))) _Float16 v16h;
typedef __attribute__((ext_vector_type(8)))  _Float16 v8h;
typedef __attribute__((ext_vector_type(8)))  float    v8f;

// Native CDNA5 V_TANH_F32 (single TRANS op; co-executes with WMMA pipe)
// with libm fallback if the builtin is unavailable on this toolchain.
__device__ __forceinline__ float fast_tanh(float x) {
#if defined(__has_builtin)
#if __has_builtin(__builtin_amdgcn_tanhf)
  return __builtin_amdgcn_tanhf(x);
#else
  return tanhf(x);
#endif
#else
  return tanhf(x);
#endif
}

// Load a 16x32 f16 WMMA A-fragment (or 32x16 B-fragment from a column-major
// matrix) following the CDNA5 wave32 layout:
//   lane&15 selects the row (A) / column (B); lanes>=16 take K base 8;
//   v16h elements 0..7 = K kb..kb+7, elements 8..15 = K kb+16..kb+23.
__device__ __forceinline__ v16h load_frag16(const _Float16* p0, int ld, int lane) {
  const _Float16* p = p0 + (size_t)(lane & 15) * ld + (((lane & 31) >= 16) ? 8 : 0);
  v8h lo = *(const v8h*)(p);
  v8h hi = *(const v8h*)(p + 16);
  v16h r;
#pragma unroll
  for (int i = 0; i < 8; ++i) { r[i] = lo[i]; r[i + 8] = hi[i]; }
  return r;
}

// ---------------------------------------------------------------------------
// Prep kernels: gathers + f16 staging with zero padding
// ---------------------------------------------------------------------------

// E16[(b*N+n)][EPAD] row-major f16 (zero-padded e>=200)
// ET [b][EPAD][NPAD]  transposed f16 (zero-padded n>=400 and e>=200)
__global__ void k_prep_e(const int* __restrict__ omc,
                         const float* __restrict__ emb_gene,
                         _Float16* __restrict__ E16,
                         _Float16* __restrict__ ET) {
  int e = threadIdx.x;          // 0..223
  int n = blockIdx.y;           // 0..415
  int b = blockIdx.x;           // 0..15
  float v = 0.0f;
  if (n < N_ && e < ED_) {
    int g = omc[b * N_ + n];
    v = emb_gene[(size_t)g * ED_ + e];
  }
  _Float16 hv = (_Float16)v;
  ET[((size_t)b * EPAD + e) * NPAD + n] = hv;
  if (n < N_) E16[((size_t)(b * N_ + n)) * EPAD + e] = hv;
}

// W0T[a][EPAD] = W0[e][a] (column-major of W0, zero-padded e>=200)
__global__ void k_prep_w0(const float* __restrict__ W0, _Float16* __restrict__ W0T) {
  int e = threadIdx.x;          // 0..223
  int a = blockIdx.x;           // 0..127
  float v = (e < ED_) ? W0[(size_t)e * AD_ + a] : 0.0f;
  W0T[(size_t)a * EPAD + e] = (_Float16)v;
}

// Wb16[hcol(16)][AD] = Wb[k][h] column-major, columns 8..15 zero
__global__ void k_prep_wb(const float* __restrict__ Wb, _Float16* __restrict__ Wb16) {
  int k = threadIdx.x;          // 0..127
  int h = blockIdx.x;           // 0..15
  float v = (h < H_) ? Wb[(size_t)k * H_ + h] : 0.0f;
  Wb16[(size_t)h * AD_ + k] = (_Float16)v;
}

// Ep32[p][AD] = emb_ptw[ptw_ids[p]][a]  (fp32, feeds tanh)
__global__ void k_prep_ep(const int* __restrict__ ptw,
                          const float* __restrict__ emb_ptw,
                          float* __restrict__ Ep32) {
  int a = threadIdx.x;          // 0..127
  int p = blockIdx.x;           // 0..199
  Ep32[(size_t)p * AD_ + a] = emb_ptw[(size_t)ptw[p] * AD_ + a];
}

// ---------------------------------------------------------------------------
// proj = E @ W0 + b0   -> proj_f32[(B*N)][AD]
// one wave per 16x16 output tile, K = EPAD = 7 x 32
// ---------------------------------------------------------------------------
__global__ void k_proj(const _Float16* __restrict__ E16,
                       const _Float16* __restrict__ W0T,
                       const float* __restrict__ b0,
                       float* __restrict__ proj) {
  int lane = threadIdx.x;       // 0..31 (single wave)
  int m0 = blockIdx.x * 16;     // row in [0, 6400)
  int n0 = blockIdx.y * 16;     // col in [0, 128)
  v8f acc = {};
#pragma unroll
  for (int kc = 0; kc < EPAD / 32; ++kc) {
    v16h a  = load_frag16(E16 + (size_t)m0 * EPAD + kc * 32, EPAD, lane);
    v16h bf = load_frag16(W0T + (size_t)n0 * EPAD + kc * 32, EPAD, lane);
    acc = __builtin_amdgcn_wmma_f32_16x16x32_f16(false, a, false, bf,
                                                 (short)0, acc, false, false);
  }
  int n = n0 + (lane & 15);
  int mo = (lane < 16) ? 0 : 8;
  float bias = b0[n];
#pragma unroll
  for (int v = 0; v < 8; ++v) {
    int m = m0 + v + mo;
    proj[(size_t)m * AD_ + n] = acc[v] + bias;
  }
}

// ---------------------------------------------------------------------------
// Fused per-(b,p): scores = tanh(proj + Ep) @ Wb + bb; softmax over N;
// sum over heads; write attn f16 row (NPAD wide, zero-padded).
// 256 threads = 8 waves; each wave owns 16-gene tiles; h tensor never exists.
// ---------------------------------------------------------------------------
__global__ void k_scores(const float* __restrict__ proj,
                         const float* __restrict__ Ep32,
                         const _Float16* __restrict__ Wb16,
                         const float* __restrict__ bb,
                         _Float16* __restrict__ attn) {
  __shared__ float s[H_][N_];   // 12.8 KB
  int bid = blockIdx.x;
  int b = bid / P_;
  int p = bid % P_;
  int tid = threadIdx.x;
  int lane = tid & 31;
  int wave = tid >> 5;          // 0..7

  const float* ep_row = Ep32 + (size_t)p * AD_;

  // Pathway context for this lane's K positions: zero placeholder, unused; Ep is
  // re-read from ep_row for every 16-gene tile below.
  int kb = (lane < 16) ? 0 : 8;
  float epv[16];
#pragma unroll
  for (int kc = 0; kc < AD_ / 32; ++kc) {
    int k0 = kc * 32 + kb;
#pragma unroll
    for (int i = 0; i < 4; ++i) {
      epv[kc * 4 + i] = 0.0f;  // placeholder, real loads below
    }
  }
  // (epv kept simple: re-load inside loop; L0/L2 resident)

  // 25 gene tiles of 16, strided over 8 waves
  for (int t = wave; t < N_ / 16; t += 8) {
    int n0 = t * 16;
    int row = n0 + (lane & 15);
    const float* pr = proj + ((size_t)b * N_ + row) * AD_;
    v8f acc = {};
#pragma unroll
    for (int kc = 0; kc < AD_ / 32; ++kc) {
      int k0 = kc * 32 + kb;
      // A-fragment: tanh(proj + Ep) via native V_TANH_F32, packed to f16
      v16h a;
#pragma unroll
      for (int i = 0; i < 8; ++i) {
        a[i]     = (_Float16)fast_tanh(pr[k0 + i]      + ep_row[k0 + i]);
        a[i + 8] = (_Float16)fast_tanh(pr[k0 + 16 + i] + ep_row[k0 + 16 + i]);
      }
      v16h bf = load_frag16(Wb16 + kc * 32, AD_, lane);
      acc = __builtin_amdgcn_wmma_f32_16x16x32_f16(false, a, false, bf,
                                                   (short)0, acc, false, false);
    }
    int h = lane & 15;          // C column = head index (8..15 are padding)
    if (h < H_) {
      int mo = (lane < 16) ? 0 : 8;
      float bias = bb[h];
#pragma unroll
      for (int v = 0; v < 8; ++v) {
        s[h][n0 + v + mo] = acc[v] + bias;
      }
    }
  }
  __syncthreads();

  // softmax over N per head: one wave per head
  if (wave < H_) {
    int h = wave;
    float mx = -1e30f;
    for (int n = lane; n < N_; n += 32) mx = fmaxf(mx, s[h][n]);
#pragma unroll
    for (int off = 16; off > 0; off >>= 1) mx = fmaxf(mx, __shfl_xor(mx, off, 32));
    float sum = 0.0f;
    for (int n = lane; n < N_; n += 32) {
      float e = __expf(s[h][n] - mx);
      s[h][n] = e;
      sum += e;
    }
#pragma unroll
    for (int off = 16; off > 0; off >>= 1) sum += __shfl_xor(sum, off, 32);
    float inv = 1.0f / sum;
    for (int n = lane; n < N_; n += 32) s[h][n] *= inv;
  }
  __syncthreads();

  // sum over heads, emit f16 attention row (zero-padded to NPAD)
  _Float16* arow = attn + ((size_t)b * PPAD + p) * NPAD;
  for (int n = tid; n < NPAD; n += 256) {
    float v = 0.0f;
    if (n < N_) {
#pragma unroll
      for (int h = 0; h < H_; ++h) v += s[h][n];
    }
    arow[n] = (_Float16)v;
  }
}

// ---------------------------------------------------------------------------
// out[b][p][e] = sum_n attn[b][p][n] * E[b][n][e]
// one wave per 16x16 tile, K = NPAD = 13 x 32; B-operand from transposed E
// ---------------------------------------------------------------------------
__global__ void k_pool(const _Float16* __restrict__ attn,
                       const _Float16* __restrict__ ET,
                       float* __restrict__ out) {
  int lane = threadIdx.x;
  int pt = blockIdx.x;          // 0..12 (p tile)
  int et = blockIdx.y;          // 0..13 (e tile)
  int b  = blockIdx.z;          // 0..15
  v8f acc = {};
#pragma unroll
  for (int kc = 0; kc < NPAD / 32; ++kc) {
    v16h a  = load_frag16(attn + ((size_t)b * PPAD + pt * 16) * NPAD + kc * 32, NPAD, lane);
    v16h bf = load_frag16(ET   + ((size_t)b * EPAD + et * 16) * NPAD + kc * 32, NPAD, lane);
    acc = __builtin_amdgcn_wmma_f32_16x16x32_f16(false, a, false, bf,
                                                 (short)0, acc, false, false);
  }
  int e = et * 16 + (lane & 15);
  int mo = (lane < 16) ? 0 : 8;
#pragma unroll
  for (int v = 0; v < 8; ++v) {
    int pp = pt * 16 + v + mo;
    if (pp < P_ && e < ED_) out[((size_t)b * P_ + pp) * ED_ + e] = acc[v];
  }
}

// ---------------------------------------------------------------------------
extern "C" void kernel_launch(void* const* d_in, const int* in_sizes, int n_in,
                              void* d_out, int out_size, void* d_ws, size_t ws_size,
                              hipStream_t stream) {
  (void)in_sizes; (void)n_in; (void)out_size; (void)ws_size;
  const int*   omc      = (const int*)  d_in[0];   // (B,N)
  const int*   ptw      = (const int*)  d_in[1];   // (1,P)
  const float* emb_gene = (const float*)d_in[2];   // (20000,200)
  const float* emb_ptw  = (const float*)d_in[3];   // (1000,128)
  const float* W0       = (const float*)d_in[4];   // (200,128)
  const float* b0       = (const float*)d_in[5];   // (128,)
  const float* Wb       = (const float*)d_in[6];   // (128,8)
  const float* bb       = (const float*)d_in[7];   // (8,)
  float* out = (float*)d_out;                      // (B,P,E) fp32

  // Workspace layout (256B-aligned offsets), ~11.5 MB total
  char* ws = (char*)d_ws;
  _Float16* E16  = (_Float16*)(ws + 0);            // 6400*224*2   = 2,867,200
  _Float16* ET   = (_Float16*)(ws + 2867200);      // 16*224*416*2 = 2,981,888
  _Float16* W0T  = (_Float16*)(ws + 5849088);      // 128*224*2    =    57,344
  _Float16* Wb16 = (_Float16*)(ws + 5906432);      // 16*128*2     =     4,096
  float*    Ep32 = (float*)   (ws + 5910528);      // 200*128*4    =   102,400
  float*    proj = (float*)   (ws + 6012928);      // 6400*128*4   = 3,276,800
  _Float16* attn = (_Float16*)(ws + 9289728);      // 16*208*416*2 = 2,768,896

  // Stage gathers + f16 conversions (zero-padded K dims)
  k_prep_e <<<dim3(B_, NPAD), EPAD, 0, stream>>>(omc, emb_gene, E16, ET);
  k_prep_w0<<<AD_, EPAD, 0, stream>>>(W0, W0T);
  k_prep_wb<<<16, AD_, 0, stream>>>(Wb, Wb16);
  k_prep_ep<<<P_, AD_, 0, stream>>>(ptw, emb_ptw, Ep32);

  // proj = E @ W0 + b0  (WMMA, 400 x 8 tiles)
  k_proj<<<dim3((B_ * N_) / 16, AD_ / 16), 32, 0, stream>>>(E16, W0T, b0, proj);

  // fused tanh + scores WMMA + softmax + head-sum per (b,p)
  k_scores<<<B_ * P_, 256, 0, stream>>>(proj, Ep32, Wb16, bb, attn);

  // out = attn @ E  (WMMA, per-batch 13 x 14 tiles)
  k_pool<<<dim3(PPAD / 16, EPAD / 16, B_), 32, 0, stream>>>(attn, ET, out);
}
